// TriangleMultiplicationOutgoing_1898375545292
// MI455X (gfx1250) — matrix operand order repaired and checked by reading, not compiled, not run on hardware
//
#include <hip/hip_runtime.h>

// ---------------------------------------------------------------------------
// TriangleMultiplicationOutgoing  (B=1, N=512, D=128)  — CDNA5 / gfx1250
// Pipeline:
//   k_prep   : fp32 weights -> bf16 (Wc = [p_in;g_in;g_out] 640x128, Wout 128x128)
//   k_stage1 : LN(x) -> h = (x W_p^T)*sigmoid(x W_g^T)*mask  (bf16 WMMA, transposed
//              orientation so stores land in A2[d][i*512+k] / B2[d][j*512+k]),
//              plus gate = sigmoid(x W_gout^T) stored [m][128]
//   k_stage2 : t[d][i][j] = sum_k A2[d][i][k] * B2[d][j][k]  (128 batched GEMMs),
//              operands staged into LDS via global_load_async_to_lds_b128
//   k_stage3 : LN(t) GEMM with p_out (bf16 WMMA) * gate -> out [m][128] fp32
// ---------------------------------------------------------------------------

#define NSZ 512
#define NN (512 * 512)
#define DSZ 128

typedef __bf16 bf16_t;
typedef __attribute__((ext_vector_type(16))) __bf16 v16bf;
typedef __attribute__((ext_vector_type(8))) float v8f;
typedef __attribute__((ext_vector_type(4))) int v4i_t;

#if defined(__has_builtin)
#if __has_builtin(__builtin_amdgcn_global_load_async_to_lds_b128)
#define HAVE_ASYNC_LDS 1
#endif
#endif

// exact parameter types per clang diagnostic: AS1 (global) int4*, AS3 (LDS) int4*
typedef __attribute__((address_space(1))) v4i_t* gp_v4i;
typedef __attribute__((address_space(3))) v4i_t* lp_v4i;

union frag_u {
    v16bf v;
    uint4 q[2];
    unsigned short u[16];
};

__device__ __forceinline__ unsigned short f2bf(float f) {
    unsigned int u = __float_as_uint(f);
    unsigned int r = u + 0x7FFFu + ((u >> 16) & 1u);
    return (unsigned short)(r >> 16);
}
__device__ __forceinline__ float bf2f(unsigned short h) {
    return __uint_as_float(((unsigned int)h) << 16);
}
__device__ __forceinline__ float sigm(float x) {
#if defined(__has_builtin)
#if __has_builtin(__builtin_amdgcn_rcpf)
    return __builtin_amdgcn_rcpf(1.0f + __expf(-x));  // v_rcp_f32, no div chain
#else
    return 1.0f / (1.0f + __expf(-x));
#endif
#else
    return 1.0f / (1.0f + __expf(-x));
#endif
}
__device__ __forceinline__ float wred(float v) {
    #pragma unroll
    for (int o = 16; o > 0; o >>= 1) v += __shfl_xor(v, o, 32);
    return v;
}
__device__ __forceinline__ uint4 ld16(const unsigned short* p) {
    return *(const uint4*)p;
}
__device__ __forceinline__ v8f wmma_bf16(v16bf a, v16bf b, v8f c) {
    return __builtin_amdgcn_wmma_f32_16x16x32_bf16(false, a, false, b, (short)0, c,
                                                   false, false);
}
__device__ __forceinline__ void wait_async0() {
#ifdef HAVE_ASYNC_LDS
#if __has_builtin(__builtin_amdgcn_s_wait_asynccnt)
    __builtin_amdgcn_s_wait_asynccnt(0);
#else
    asm volatile("s_wait_asynccnt 0" ::: "memory");
#endif
#endif
}

// ---------------------------------------------------------------------------
// k_prep: convert weights fp32 -> bf16.  Wc rows: 0..255 p_in, 256..511 g_in,
// 512..639 g_out.  Wout = p_out (128x128).
// ---------------------------------------------------------------------------
__global__ __launch_bounds__(256) void k_prep(const float* __restrict__ p_in,
                                              const float* __restrict__ g_in,
                                              const float* __restrict__ g_out,
                                              const float* __restrict__ p_out,
                                              unsigned short* __restrict__ Wc,
                                              unsigned short* __restrict__ Wout) {
    int i = blockIdx.x * 256 + threadIdx.x;
    if (i < 256 * 128)
        Wc[i] = f2bf(p_in[i]);
    else if (i < 512 * 128)
        Wc[i] = f2bf(g_in[i - 256 * 128]);
    else if (i < 640 * 128)
        Wc[i] = f2bf(g_out[i - 512 * 128]);
    if (i < 128 * 128) Wout[i] = f2bf(p_out[i]);
}

// ---------------------------------------------------------------------------
// k_stage1: 64 rows (m) per block, 256 threads = 8 waves.
// ---------------------------------------------------------------------------
__global__ __launch_bounds__(256) void k_stage1(
    const float* __restrict__ x, const float* __restrict__ mask,
    const float* __restrict__ nw, const float* __restrict__ nb,
    const unsigned short* __restrict__ Wc,  // [640][128] bf16
    unsigned short* __restrict__ A2,        // [128][NN]  a, k-contiguous
    unsigned short* __restrict__ B2,        // [128][NN]  b, k-contiguous
    unsigned short* __restrict__ gateb)     // [NN][128]  sigmoid gate (bf16)
{
    __shared__ unsigned short ln_s[64][136];  // padded rows (16B aligned)

    const int tid = threadIdx.x;
    const int wave = tid >> 5, lane = tid & 31;
    const long m0 = (long)blockIdx.x * 64;

    // ---- LayerNorm: each wave handles 8 rows, 32 lanes x float4 per row ----
    #pragma unroll 2
    for (int rr = 0; rr < 8; ++rr) {
        int row = wave * 8 + rr;
        const float4 v = ((const float4*)(x + (m0 + row) * DSZ))[lane];
        float s = wred(v.x + v.y + v.z + v.w);
        float mu = s * (1.0f / 128.0f);
        float d0 = v.x - mu, d1 = v.y - mu, d2 = v.z - mu, d3 = v.w - mu;
        float ss = wred(d0 * d0 + d1 * d1 + d2 * d2 + d3 * d3);
        float rstd = rsqrtf(ss * (1.0f / 128.0f) + 1e-5f);
        float4 w4 = ((const float4*)nw)[lane];
        float4 b4 = ((const float4*)nb)[lane];
        uint2 pk;
        pk.x = (unsigned)f2bf(d0 * rstd * w4.x + b4.x) |
               ((unsigned)f2bf(d1 * rstd * w4.y + b4.y) << 16);
        pk.y = (unsigned)f2bf(d2 * rstd * w4.z + b4.z) |
               ((unsigned)f2bf(d3 * rstd * w4.w + b4.w) << 16);
        *(uint2*)&ln_s[row][lane * 4] = pk;
    }
    __syncthreads();

    // ---- GEMM jobs: 4 m-tiles x (16 h-jobs + 8 gate-jobs) = 96 jobs --------
    const int lhalf = lane & 15;
    const int khiA = (lane >> 4) << 4;   // A frag: +16 K for hi half-wave
    const int khiB = (lane >> 4) << 4;   // B frag: +16 K for hi half-wave
    const int rhi = (lane >> 4) * 8;

    for (int j = wave; j < 96; j += 8) {
        const int mt = j / 24;
        const int r = j % 24;
        const int mrow0 = mt * 16;
        const long mglob = m0 + mrow0;

        if (r < 16) {
            // h-job, TRANSPOSED orientation: A = Wc rows (e), B = ln cols (m)
            const int ep = r * 16;        // p rows ep..ep+15
            const int eg = 256 + r * 16;  // paired g rows
            const unsigned short* wp = Wc + (ep + lhalf) * DSZ;
            const unsigned short* wg = Wc + (eg + lhalf) * DSZ;
            const unsigned short* lm = &ln_s[mrow0 + lhalf][0];
            v8f cp = {0.f, 0.f, 0.f, 0.f, 0.f, 0.f, 0.f, 0.f};
            v8f cg = {0.f, 0.f, 0.f, 0.f, 0.f, 0.f, 0.f, 0.f};
            #pragma unroll
            for (int k0 = 0; k0 < 128; k0 += 32) {
                frag_u ap, ag, bm;
                ap.q[0] = ld16(wp + k0 + khiA);
                ap.q[1] = ld16(wp + k0 + khiA + 8);
                ag.q[0] = ld16(wg + k0 + khiA);
                ag.q[1] = ld16(wg + k0 + khiA + 8);
                bm.q[0] = *(const uint4*)(lm + k0 + khiB);
                bm.q[1] = *(const uint4*)(lm + k0 + khiB + 8);
                cp = wmma_bf16(ap.v, bm.v, cp);
                cg = wmma_bf16(ag.v, bm.v, cg);
            }
            // C rows = e, C cols = m  ->  scatter into A2/B2 (m contiguous)
            const float mk = mask[mglob + lhalf];
            unsigned short* dst =
                (r < 8) ? (A2 + (long)ep * NN) : (B2 + (long)(ep - 128) * NN);
            #pragma unroll
            for (int rr = 0; rr < 8; ++rr) {
                float hv = cp[rr] * sigm(cg[rr]) * mk;
                dst[(long)(rr + rhi) * NN + mglob + lhalf] = f2bf(hv);
            }
        } else {
            // gate-job, NORMAL orientation: A = ln rows (m), B = Wc rows (c)
            const int c0 = 512 + (r - 16) * 16;
            const unsigned short* la = &ln_s[mrow0 + lhalf][0];
            const unsigned short* wc = Wc + (c0 + lhalf) * DSZ;
            v8f cc = {0.f, 0.f, 0.f, 0.f, 0.f, 0.f, 0.f, 0.f};
            #pragma unroll
            for (int k0 = 0; k0 < 128; k0 += 32) {
                frag_u am, bw;
                am.q[0] = *(const uint4*)(la + k0 + khiA);
                am.q[1] = *(const uint4*)(la + k0 + khiA + 8);
                bw.q[0] = ld16(wc + k0 + khiB);
                bw.q[1] = ld16(wc + k0 + khiB + 8);
                cc = wmma_bf16(am.v, bw.v, cc);
            }
            // C rows = m, C cols = c  ->  gate[m][c]
            #pragma unroll
            for (int rr = 0; rr < 8; ++rr) {
                long m = mglob + rr + rhi;
                gateb[m * DSZ + (c0 - 512) + lhalf] = f2bf(sigm(cc[rr]));
            }
        }
    }
}

// ---------------------------------------------------------------------------
// k_stage2: t[d][i][j] = sum_k A2[d][i][k]*B2[d][j][k].  One (d, 64x64) tile
// per block.  64x128 K-chunks of A and B staged into LDS with async-to-LDS
// copies (ASYNCcnt path); each wave owns one 16-row i-tile x 32-wide j-pair.
// ---------------------------------------------------------------------------
__global__ __launch_bounds__(256) void k_stage2(const unsigned short* __restrict__ A2,
                                                const unsigned short* __restrict__ B2,
                                                float* __restrict__ t) {
    __shared__ unsigned short As[64][136];  // 64 rows x 128 K, padded (+16B)
    __shared__ unsigned short Bs[64][136];

    const int bx = blockIdx.x;              // 128 * 8 * 8 = 8192 blocks
    const int d = bx >> 6;
    const int ib = (bx >> 3) & 7, jb = bx & 7;
    const unsigned short* Ad = A2 + (long)d * NN;
    const unsigned short* Bd = B2 + (long)d * NN;
    const int tid = threadIdx.x;
    const int wave = tid >> 5, lane = tid & 31;
    const int lhalf = lane & 15;
    const int khiA = (lane >> 4) << 4;
    const int khiB = (lane >> 4) << 4;

    // staging coords: each thread moves 4 x 16B per buffer per K-chunk
    const int srow = tid >> 4;           // rows srow, srow+16, +32, +48
    const int scol = (tid & 15) * 8;     // ushort offset of this 16B chunk

    const unsigned short* ga = Ad + (long)(ib * 64 + srow) * NSZ + scol;
    const unsigned short* gb = Bd + (long)(jb * 64 + srow) * NSZ + scol;

    const int arow = (wave & 3) * 16 + lhalf;
    const int brow0 = (wave >> 2) * 32 + lhalf;
    const int brow1 = brow0 + 16;

    v8f c0v = {0.f, 0.f, 0.f, 0.f, 0.f, 0.f, 0.f, 0.f};
    v8f c1v = {0.f, 0.f, 0.f, 0.f, 0.f, 0.f, 0.f, 0.f};

    for (int k0 = 0; k0 < 512; k0 += 128) {
        // ---- fill LDS: 2 x 16KB (A and B sub-tiles) --------------------
        #pragma unroll
        for (int n = 0; n < 4; ++n) {
            const int row = srow + n * 16;
            const long goff = (long)n * 16 * NSZ + k0;
#ifdef HAVE_ASYNC_LDS
            __builtin_amdgcn_global_load_async_to_lds_b128(
                (gp_v4i)(ga + goff), (lp_v4i)&As[row][scol], 0, 0);
            __builtin_amdgcn_global_load_async_to_lds_b128(
                (gp_v4i)(gb + goff), (lp_v4i)&Bs[row][scol], 0, 0);
#else
            *(uint4*)&As[row][scol] = ld16(ga + goff);
            *(uint4*)&Bs[row][scol] = ld16(gb + goff);
#endif
        }
        wait_async0();
        __syncthreads();

        // ---- consume: 4 WMMA K-steps per C tile ------------------------
        #pragma unroll
        for (int kk = 0; kk < 128; kk += 32) {
            frag_u a, b0, b1;
            a.q[0] = *(const uint4*)&As[arow][kk + khiA];
            a.q[1] = *(const uint4*)&As[arow][kk + khiA + 8];
            b0.q[0] = *(const uint4*)&Bs[brow0][kk + khiB];
            b0.q[1] = *(const uint4*)&Bs[brow0][kk + khiB + 8];
            b1.q[0] = *(const uint4*)&Bs[brow1][kk + khiB];
            b1.q[1] = *(const uint4*)&Bs[brow1][kk + khiB + 8];
            c0v = wmma_bf16(a.v, b0.v, c0v);
            c1v = wmma_bf16(a.v, b1.v, c1v);
        }
        __syncthreads();
    }

    float* td = t + (long)d * NN;
    const int it = ib * 64 + (wave & 3) * 16;
    const int jt = jb * 64 + (wave >> 2) * 32;
    const int rhi = (lane >> 4) * 8;
    #pragma unroll
    for (int rr = 0; rr < 8; ++rr) {
        int i = it + rr + rhi;
        td[(long)i * NSZ + jt + lhalf] = c0v[rr];
        td[(long)i * NSZ + jt + 16 + lhalf] = c1v[rr];
    }
}

// ---------------------------------------------------------------------------
// k_stage3: LN(t) over d, GEMM with Wout, multiply by stored gate, store fp32.
// 64 m-rows per block; t is stored [d][m] so gather via LDS (256B runs).
// ---------------------------------------------------------------------------
__global__ __launch_bounds__(256) void k_stage3(
    const float* __restrict__ t, const unsigned short* __restrict__ gateb,
    const float* __restrict__ nw, const float* __restrict__ nb,
    const unsigned short* __restrict__ Wout,  // [128][128] bf16
    float* __restrict__ out) {
    __shared__ float t_s[64][129];            // pad: conflict-free gather
    __shared__ unsigned short ln_s[64][136];

    const int tid = threadIdx.x;
    const int wave = tid >> 5, lane = tid & 31;
    const long m0 = (long)blockIdx.x * 64;

    // ---- cooperative gather: t[d][m0..m0+63] -> t_s[m][d] ------------------
    #pragma unroll 4
    for (int rep = 0; rep < 32; ++rep) {
        int idx = rep * 256 + tid;
        int d = idx >> 6, mm = idx & 63;
        t_s[mm][d] = t[(long)d * NN + m0 + mm];
    }
    __syncthreads();

    // ---- LayerNorm over d --------------------------------------------------
    #pragma unroll 2
    for (int rr = 0; rr < 8; ++rr) {
        int row = wave * 8 + rr;
        float v0 = t_s[row][lane * 4 + 0], v1 = t_s[row][lane * 4 + 1];
        float v2 = t_s[row][lane * 4 + 2], v3 = t_s[row][lane * 4 + 3];
        float s = wred(v0 + v1 + v2 + v3);
        float mu = s * (1.0f / 128.0f);
        float d0 = v0 - mu, d1 = v1 - mu, d2 = v2 - mu, d3 = v3 - mu;
        float ss = wred(d0 * d0 + d1 * d1 + d2 * d2 + d3 * d3);
        float rstd = rsqrtf(ss * (1.0f / 128.0f) + 1e-5f);
        float4 w4 = ((const float4*)nw)[lane];
        float4 b4 = ((const float4*)nb)[lane];
        uint2 pk;
        pk.x = (unsigned)f2bf(d0 * rstd * w4.x + b4.x) |
               ((unsigned)f2bf(d1 * rstd * w4.y + b4.y) << 16);
        pk.y = (unsigned)f2bf(d2 * rstd * w4.z + b4.z) |
               ((unsigned)f2bf(d3 * rstd * w4.w + b4.w) << 16);
        *(uint2*)&ln_s[row][lane * 4] = pk;
    }
    __syncthreads();

    // ---- GEMM: out[m][e] = (ln @ Wout^T) * gate ----------------------------
    const int lhalf = lane & 15;
    const int khiA = (lane >> 4) << 4;
    const int khiB = (lane >> 4) << 4;
    const int rhi = (lane >> 4) * 8;
    const int mrow0 = (wave & 3) * 16;
    const unsigned short* la = &ln_s[mrow0 + lhalf][0];

    #pragma unroll
    for (int eo = 0; eo < 4; ++eo) {
        const int e0 = ((wave >> 2) * 4 + eo) * 16;
        const unsigned short* wr = Wout + (e0 + lhalf) * DSZ;
        v8f c = {0.f, 0.f, 0.f, 0.f, 0.f, 0.f, 0.f, 0.f};
        #pragma unroll
        for (int k0 = 0; k0 < 128; k0 += 32) {
            frag_u a, b;
            a.q[0] = *(const uint4*)(la + k0 + khiA);
            a.q[1] = *(const uint4*)(la + k0 + khiA + 8);
            b.q[0] = ld16(wr + k0 + khiB);
            b.q[1] = ld16(wr + k0 + khiB + 8);
            c = wmma_bf16(a.v, b.v, c);
        }
        #pragma unroll
        for (int rr = 0; rr < 8; ++rr) {
            long m = m0 + mrow0 + rr + rhi;
            int e = e0 + lhalf;
            out[m * DSZ + e] = c[rr] * bf2f(gateb[m * DSZ + e]);
        }
    }
}

// ---------------------------------------------------------------------------
extern "C" void kernel_launch(void* const* d_in, const int* in_sizes, int n_in,
                              void* d_out, int out_size, void* d_ws, size_t ws_size,
                              hipStream_t stream) {
    const float* x = (const float*)d_in[0];
    const float* mask = (const float*)d_in[1];
    const float* norm_in_w = (const float*)d_in[2];
    const float* norm_in_b = (const float*)d_in[3];
    const float* p_in_w = (const float*)d_in[4];
    const float* g_in_w = (const float*)d_in[5];
    const float* norm_out_w = (const float*)d_in[6];
    const float* norm_out_b = (const float*)d_in[7];
    const float* p_out_w = (const float*)d_in[8];
    const float* g_out_w = (const float*)d_in[9];
    float* out = (float*)d_out;

    char* ws = (char*)d_ws;
    unsigned short* Wc = (unsigned short*)(ws);                       // 160 KB
    unsigned short* Wout = (unsigned short*)(ws + 163840);            //  32 KB
    unsigned short* A2 = (unsigned short*)(ws + 196608);              //  64 MB
    unsigned short* B2 = (unsigned short*)(ws + 196608 + 67108864ull);
    unsigned short* gateb = (unsigned short*)(ws + 196608 + 2ull * 67108864ull);
    float* tbuf = (float*)(ws + 196608 + 3ull * 67108864ull);         // 128 MB

    k_prep<<<320, 256, 0, stream>>>(p_in_w, g_in_w, g_out_w, p_out_w, Wc, Wout);
    k_stage1<<<NN / 64, 256, 0, stream>>>(x, mask, norm_in_w, norm_in_b, Wc, A2, B2,
                                          gateb);
    k_stage2<<<128 * 64, 256, 0, stream>>>(A2, B2, tbuf);
    k_stage3<<<NN / 64, 256, 0, stream>>>(tbuf, gateb, norm_out_w, norm_out_b, Wout,
                                          out);
    (void)in_sizes; (void)n_in; (void)out_size; (void)ws_size;
}
